// ISA_72705206386984
// MI455X (gfx1250) — matrix operand order repaired and checked by reading, not compiled, not run on hardware
//
#include <hip/hip_runtime.h>
#include <math.h>

typedef unsigned short ushort_t;
typedef unsigned int u32;
typedef __attribute__((ext_vector_type(4)))  u32     v4u;
typedef __attribute__((ext_vector_type(16))) __bf16  v16bf;
typedef __attribute__((ext_vector_type(8)))  float   v8f;

#define BB    8
#define CC    256
#define HH    64
#define WWD   64
#define HWD   (HH * WWD)
#define NPIX  (BB * HH * WWD)      // 32768
#define NHEAD 8
#define HD    32

// ---------------- bf16 helpers ----------------
static __device__ __forceinline__ ushort_t f2bf(float f) {
  u32 u = __float_as_uint(f);
  u += 0x7FFFu + ((u >> 16) & 1u);           // round-to-nearest-even
  return (ushort_t)(u >> 16);
}
static __device__ __forceinline__ float bf2f(ushort_t s) {
  return __uint_as_float(((u32)s) << 16);
}

struct Frag32B { v4u lo, hi; };
static __device__ __forceinline__ v16bf make_frag(const ushort_t* p0, const ushort_t* p1) {
  Frag32B f;
  f.lo = *(const v4u*)p0;
  f.hi = *(const v4u*)p1;
  return __builtin_bit_cast(v16bf, f);
}

// ---------------- WMMA core: one wave computes a 16(M) x 64(N) strip ----------------
// A: [M,K] bf16 row-major.  Bw: [N,K] bf16 row-major (i.e., weight W[n,k], out = A @ W^T).
// Fully-unrolled software pipeline (K compile-time): fragments for step s+1 are
// loaded before the WMMA chain of step s; full unroll renames away the ping-pong.
template <int K>
static __device__ __forceinline__ void wmma_16x64(
    const ushort_t* __restrict__ A, const ushort_t* __restrict__ Bw,
    int m0, int n0, v8f acc[4])
{
  static_assert(K % 64 == 0, "K must be a multiple of 64");
  constexpr int S = K / 32;              // even number of k-steps
  const int lane = threadIdx.x & 31;
  const int half = lane >> 4;
  const int lr   = lane & 15;
  const ushort_t* arow = A + (size_t)(m0 + lr) * K + half * 8;
  const ushort_t* brow[4];
#pragma unroll
  for (int t = 0; t < 4; ++t)
    brow[t] = Bw + (size_t)(n0 + t * 16 + lr) * K + half * 8;

  // stage 0
  v16bf a0 = make_frag(arow, arow + 16);
  v16bf b0[4];
#pragma unroll
  for (int t = 0; t < 4; ++t) b0[t] = make_frag(brow[t], brow[t] + 16);
  v16bf a1;
  v16bf b1[4];

#pragma unroll
  for (int s = 0; s < S; s += 2) {
    // prefetch step s+1 (always exists: S even)
    {
      const int kn = (s + 1) * 32;
      a1 = make_frag(arow + kn, arow + kn + 16);
#pragma unroll
      for (int t = 0; t < 4; ++t) b1[t] = make_frag(brow[t] + kn, brow[t] + kn + 16);
    }
    // compute step s
#pragma unroll
    for (int t = 0; t < 4; ++t)
      acc[t] = __builtin_amdgcn_wmma_f32_16x16x32_bf16(
          false, a0, false, b0[t], (short)0, acc[t], false, false);
    // prefetch step s+2 (compile-time predicated)
    if (s + 2 < S) {
      const int kn = (s + 2) * 32;
      a0 = make_frag(arow + kn, arow + kn + 16);
#pragma unroll
      for (int t = 0; t < 4; ++t) b0[t] = make_frag(brow[t] + kn, brow[t] + kn + 16);
    }
    // compute step s+1
#pragma unroll
    for (int t = 0; t < 4; ++t)
      acc[t] = __builtin_amdgcn_wmma_f32_16x16x32_bf16(
          false, a1, false, b1[t], (short)0, acc[t], false, false);
  }
}

// ---------------- GEMM kernels with fused epilogues ----------------

// QKV: N=1024, K=256; de-interleave column n -> (tensor j=n&3, channel c=n>>2)
__global__ __launch_bounds__(32) void gemm_qkv_k(
    const ushort_t* __restrict__ A, const ushort_t* __restrict__ Bw,
    const float* __restrict__ bias,
    ushort_t* __restrict__ qb, ushort_t* __restrict__ kb,
    ushort_t* __restrict__ vb, ushort_t* __restrict__ cb)
{
  const int m0 = blockIdx.x * 16, n0 = blockIdx.y * 64;
  const int lane = threadIdx.x & 31, half = lane >> 4, lr = lane & 15;
  v8f acc[4];
  const v8f z = {0.f,0.f,0.f,0.f,0.f,0.f,0.f,0.f};
#pragma unroll
  for (int t = 0; t < 4; ++t) acc[t] = z;
  wmma_16x64<CC>(A, Bw, m0, n0, acc);
#pragma unroll
  for (int t = 0; t < 4; ++t) {
    const int n = n0 + t * 16 + lr;
    const float bn = bias[n];
    const int j = n & 3, c = n >> 2;
#pragma unroll
    for (int r = 0; r < 8; ++r) {
      const int m = m0 + r + half * 8;
      const ushort_t bv = f2bf(acc[t][r] + bn);
      const size_t o = (size_t)m * CC + c;
      if      (j == 0) qb[o] = bv;
      else if (j == 1) kb[o] = bv;
      else if (j == 2) vb[o] = bv;
      else             cb[o] = bv;
    }
  }
}

// Gate hidden: relu, bf16 out [M,N]
template <int K>
__global__ __launch_bounds__(32) void gemm_relu_k(
    const ushort_t* __restrict__ A, const ushort_t* __restrict__ Bw,
    const float* __restrict__ bias, int N, ushort_t* __restrict__ out)
{
  const int m0 = blockIdx.x * 16, n0 = blockIdx.y * 64;
  const int lane = threadIdx.x & 31, half = lane >> 4, lr = lane & 15;
  v8f acc[4];
  const v8f z = {0.f,0.f,0.f,0.f,0.f,0.f,0.f,0.f};
#pragma unroll
  for (int t = 0; t < 4; ++t) acc[t] = z;
  wmma_16x64<K>(A, Bw, m0, n0, acc);
#pragma unroll
  for (int t = 0; t < 4; ++t) {
    const int n = n0 + t * 16 + lr;
    const float bn = bias[n];
#pragma unroll
    for (int r = 0; r < 8; ++r) {
      const int m = m0 + r + half * 8;
      out[(size_t)m * N + n] = f2bf(fmaxf(acc[t][r] + bn, 0.f));
    }
  }
}

// Gate out: sigmoid, bf16 out [M,N]
template <int K>
__global__ __launch_bounds__(32) void gemm_sigmoid_k(
    const ushort_t* __restrict__ A, const ushort_t* __restrict__ Bw,
    const float* __restrict__ bias, int N, ushort_t* __restrict__ out)
{
  const int m0 = blockIdx.x * 16, n0 = blockIdx.y * 64;
  const int lane = threadIdx.x & 31, half = lane >> 4, lr = lane & 15;
  v8f acc[4];
  const v8f z = {0.f,0.f,0.f,0.f,0.f,0.f,0.f,0.f};
#pragma unroll
  for (int t = 0; t < 4; ++t) acc[t] = z;
  wmma_16x64<K>(A, Bw, m0, n0, acc);
#pragma unroll
  for (int t = 0; t < 4; ++t) {
    const int n = n0 + t * 16 + lr;
    const float bn = bias[n];
#pragma unroll
    for (int r = 0; r < 8; ++r) {
      const int m = m0 + r + half * 8;
      const float v = acc[t][r] + bn;
      out[(size_t)m * N + n] = f2bf(1.f / (1.f + __expf(-v)));
    }
  }
}

// Output projection: N=K=256, f32 out scattered to NCHW
__global__ __launch_bounds__(32) void gemm_proj_k(
    const ushort_t* __restrict__ A, const ushort_t* __restrict__ Bw,
    const float* __restrict__ bias, float* __restrict__ out)
{
  const int m0 = blockIdx.x * 16, n0 = blockIdx.y * 64;
  const int lane = threadIdx.x & 31, half = lane >> 4, lr = lane & 15;
  v8f acc[4];
  const v8f z = {0.f,0.f,0.f,0.f,0.f,0.f,0.f,0.f};
#pragma unroll
  for (int t = 0; t < 4; ++t) acc[t] = z;
  wmma_16x64<CC>(A, Bw, m0, n0, acc);
#pragma unroll
  for (int t = 0; t < 4; ++t) {
    const int n = n0 + t * 16 + lr;
    const float bn = bias[n];
#pragma unroll
    for (int r = 0; r < 8; ++r) {
      const int m = m0 + r + half * 8;       // pixel index
      const int b = m / HWD, rem = m % HWD;  // NCHW scatter
      out[((size_t)b * CC + n) * HWD + rem] = acc[t][r] + bn;
    }
  }
}

// ---------------- avg-pool 3x3 (count_include_pad) + NCHW -> pixel-major bf16 ----------------
__global__ __launch_bounds__(256) void avgpool_k(
    const float* __restrict__ x, ushort_t* __restrict__ yb)
{
  const int idx = blockIdx.x * 256 + threadIdx.x;
  if (idx >= BB * CC * HH * WWD) return;
  const int w = idx % WWD;
  const int h = (idx / WWD) % HH;
  const int c = (idx / (WWD * HH)) % CC;
  const int b = idx / (WWD * HH * CC);
  const float* xp = x + (size_t)(b * CC + c) * HWD;
  float s = 0.f;
#pragma unroll
  for (int dy = -1; dy <= 1; ++dy) {
    const int hh = h + dy;
    if (hh < 0 || hh >= HH) continue;
#pragma unroll
    for (int dx = -1; dx <= 1; ++dx) {
      const int ww = w + dx;
      if (ww < 0 || ww >= WWD) continue;
      s += xp[hh * WWD + ww];
    }
  }
  yb[((size_t)(b * HH + h) * WWD + w) * CC + c] = f2bf(s * (1.f / 9.f));
}

// ---------------- f32 -> bf16 convert (weights) ----------------
__global__ __launch_bounds__(256) void cvt_k(
    const float* __restrict__ in, ushort_t* __restrict__ out, int n)
{
  const int i = blockIdx.x * 256 + threadIdx.x;
  if (i < n) out[i] = f2bf(in[i]);
}

// ---------------- 3x3 local attention + gate, one thread per (pixel, head) ----------------
__global__ __launch_bounds__(256) void attn_k(
    const ushort_t* __restrict__ qb, const ushort_t* __restrict__ kb,
    const ushort_t* __restrict__ vb, const ushort_t* __restrict__ gb,
    ushort_t* __restrict__ ao)
{
  const int t = blockIdx.x * 256 + threadIdx.x;
  if (t >= NPIX * NHEAD) return;
  const int head = t & (NHEAD - 1);
  const int pix  = t >> 3;
  const int wq = pix % WWD;
  const int hq = (pix / WWD) % HH;
  const int b  = pix / HWD;
  const size_t base = (size_t)pix * CC + head * HD;

  float qv[HD];
#pragma unroll
  for (int d = 0; d < HD; ++d) qv[d] = bf2f(qb[base + d]);

  float logit[9];
  int i = 0;
  for (int dy = -1; dy <= 1; ++dy) {
    for (int dx = -1; dx <= 1; ++dx) {
      const int hh = hq + dy, ww = wq + dx;
      float dot = 0.f;
      if (hh >= 0 && hh < HH && ww >= 0 && ww < WWD) {
        const size_t nb = ((size_t)((b * HH + hh) * WWD + ww)) * CC + head * HD;
#pragma unroll
        for (int d = 0; d < HD; ++d) dot += qv[d] * bf2f(kb[nb + d]);
      }
      logit[i++] = dot * 0.17677669529663687f;   // hd^-0.5, zero-padded taps keep logit 0
    }
  }
  float mx = logit[0];
#pragma unroll
  for (int j = 1; j < 9; ++j) mx = fmaxf(mx, logit[j]);
  float wg[9], sum = 0.f;
#pragma unroll
  for (int j = 0; j < 9; ++j) { wg[j] = __expf(logit[j] - mx); sum += wg[j]; }
  const float inv = 1.f / sum;

  float outv[HD];
#pragma unroll
  for (int d = 0; d < HD; ++d) outv[d] = 0.f;
  i = 0;
  for (int dy = -1; dy <= 1; ++dy) {
    for (int dx = -1; dx <= 1; ++dx) {
      const int hh = hq + dy, ww = wq + dx;
      if (hh >= 0 && hh < HH && ww >= 0 && ww < WWD) {
        const size_t nb = ((size_t)((b * HH + hh) * WWD + ww)) * CC + head * HD;
        const float wv = wg[i] * inv;
#pragma unroll
        for (int d = 0; d < HD; ++d) outv[d] += wv * bf2f(vb[nb + d]);
      }
      ++i;   // OOB taps contribute weight mass but zero value
    }
  }
#pragma unroll
  for (int d = 0; d < HD; ++d) {
    const float g = bf2f(gb[base + d]);
    ao[base + d] = f2bf(outv[d] * (1.f + g));   // out*g + out
  }
}

// ---------------- host orchestration ----------------
extern "C" void kernel_launch(void* const* d_in, const int* in_sizes, int n_in,
                              void* d_out, int out_size, void* d_ws, size_t ws_size,
                              hipStream_t stream)
{
  const float* x    = (const float*)d_in[0];
  const float* Wqkv = (const float*)d_in[1];
  const float* bqkv = (const float*)d_in[2];
  const float* W1   = (const float*)d_in[3];
  const float* b1   = (const float*)d_in[4];
  const float* W2   = (const float*)d_in[5];
  const float* b2   = (const float*)d_in[6];
  const float* Wp   = (const float*)d_in[7];
  const float* bp   = (const float*)d_in[8];
  float* out = (float*)d_out;

  char* ws = (char*)d_ws;
  size_t off = 0;
  auto carve = [&](size_t bytes) -> char* {
    char* p = ws + off;
    off += (bytes + 255) & ~(size_t)255;
    return p;
  };
  ushort_t* yb    = (ushort_t*)carve((size_t)NPIX * CC * 2);          // pooled, pixel-major
  ushort_t* Wqkvb = (ushort_t*)carve((size_t)4 * CC * CC * 2);
  ushort_t* W1b   = (ushort_t*)carve((size_t)(CC / 2) * CC * 2);
  ushort_t* W2b   = (ushort_t*)carve((size_t)CC * (CC / 2) * 2);
  ushort_t* Wpb   = (ushort_t*)carve((size_t)CC * CC * 2);
  ushort_t* qb    = (ushort_t*)carve((size_t)NPIX * CC * 2);
  ushort_t* kb    = (ushort_t*)carve((size_t)NPIX * CC * 2);
  ushort_t* vb    = (ushort_t*)carve((size_t)NPIX * CC * 2);
  ushort_t* cb    = (ushort_t*)carve((size_t)NPIX * CC * 2);          // chans (gate input)
  ushort_t* hb    = (ushort_t*)carve((size_t)NPIX * (CC / 2) * 2);    // gate hidden
  ushort_t* gb    = (ushort_t*)carve((size_t)NPIX * CC * 2);          // gate
  ushort_t* aob   = (ushort_t*)carve((size_t)NPIX * CC * 2);          // gated attn out
  (void)ws_size; (void)in_sizes; (void)n_in; (void)out_size;

  // weight conversions
  cvt_k<<<(4 * CC * CC + 255) / 256, 256, 0, stream>>>(Wqkv, Wqkvb, 4 * CC * CC);
  cvt_k<<<((CC / 2) * CC + 255) / 256, 256, 0, stream>>>(W1, W1b, (CC / 2) * CC);
  cvt_k<<<(CC * (CC / 2) + 255) / 256, 256, 0, stream>>>(W2, W2b, CC * (CC / 2));
  cvt_k<<<(CC * CC + 255) / 256, 256, 0, stream>>>(Wp, Wpb, CC * CC);

  // pool + layout change
  avgpool_k<<<(BB * CC * HH * WWD + 255) / 256, 256, 0, stream>>>(x, yb);

  // QKV projection (M=32768, N=1024, K=256), fused de-interleave
  gemm_qkv_k<<<dim3(NPIX / 16, (4 * CC) / 64), 32, 0, stream>>>(yb, Wqkvb, bqkv, qb, kb, vb, cb);

  // gate MLP: 256 -> 128 (relu), 128 -> 256 (sigmoid)
  gemm_relu_k<CC><<<dim3(NPIX / 16, (CC / 2) / 64), 32, 0, stream>>>(cb, W1b, b1, CC / 2, hb);
  gemm_sigmoid_k<CC / 2><<<dim3(NPIX / 16, CC / 64), 32, 0, stream>>>(hb, W2b, b2, CC, gb);

  // 3x3 window attention + (1+g) gate
  attn_k<<<(NPIX * NHEAD + 255) / 256, 256, 0, stream>>>(qb, kb, vb, gb, aob);

  // output projection + NCHW transpose
  gemm_proj_k<<<dim3(NPIX / 16, CC / 64), 32, 0, stream>>>(aob, Wpb, bp, out);
}
